// CNNEmbedding_42563125903822
// MI455X (gfx1250) — matrix operand order repaired and checked by reading, not compiled, not run on hardware
//
#include <hip/hip_runtime.h>
#include <hip/hip_bf16.h>

// ---------------------------------------------------------------------------
// Problem constants (fixed by the reference): vocab=8192, inter=4352, emb=512,
// N=8192.  out = gelu(bn2(gelu(bn1(W1.T[x]+b1)) @ W2.T + b2))
// ---------------------------------------------------------------------------
#define VOCAB 8192
#define INTER 4352
#define EMB   512
#define NB    8192
#define BN_EPS 1e-5f

typedef __attribute__((ext_vector_type(8)))  float  v8f;
typedef __attribute__((ext_vector_type(16))) __bf16 v16bf;
typedef __attribute__((ext_vector_type(8)))  __bf16 v8bf;

__device__ __forceinline__ float gelu_exact(float x) {
    return 0.5f * x * (1.0f + erff(x * 0.70710678118654752f));
}

// ---------------------------------------------------------------------------
// K0: zero histogram
// ---------------------------------------------------------------------------
__global__ void k_zero(unsigned int* __restrict__ cnt) {
    int i = blockIdx.x * blockDim.x + threadIdx.x;
    if (i < VOCAB) cnt[i] = 0u;
}

// K1: histogram of token indices (integer atomics -> deterministic)
__global__ void k_hist(const int* __restrict__ x, unsigned int* __restrict__ cnt) {
    int i = blockIdx.x * blockDim.x + threadIdx.x;
    if (i < NB) atomicAdd(&cnt[x[i]], 1u);
}

// ---------------------------------------------------------------------------
// K2: BN1 stats per channel via count-weighted coalesced stream over W1.
//     mean_c = S1/N + b1[c];  var_c = S2/N - (S1/N)^2  (shift-invariant)
//     Fold BN into scale1/shift1.
// ---------------------------------------------------------------------------
__global__ __launch_bounds__(256) void k_stats1(
    const float* __restrict__ W1, const unsigned int* __restrict__ cnt,
    const float* __restrict__ b1, const float* __restrict__ g1,
    const float* __restrict__ beta1,
    float* __restrict__ scale1, float* __restrict__ shift1)
{
    const int c   = blockIdx.x;          // 0..INTER-1
    const int tid = threadIdx.x;
    const float* row = W1 + (size_t)c * VOCAB;
    float s1 = 0.f, s2 = 0.f;
    for (int v = tid; v < VOCAB; v += 256) {
        float w = row[v];
        float k = (float)cnt[v];
        s1 += k * w;
        s2 += k * w * w;
    }
    __shared__ float r1[256], r2[256];
    r1[tid] = s1; r2[tid] = s2;
    __syncthreads();
    for (int off = 128; off > 0; off >>= 1) {
        if (tid < off) { r1[tid] += r1[tid + off]; r2[tid] += r2[tid + off]; }
        __syncthreads();
    }
    if (tid == 0) {
        float invN = 1.0f / (float)NB;
        float mw   = r1[0] * invN;                  // mean of gathered weights
        float var  = r2[0] * invN - mw * mw;        // b1 shift cancels in var
        float sc   = g1[c] * rsqrtf(var + BN_EPS);
        scale1[c]  = sc;
        shift1[c]  = beta1[c] - (mw + b1[c]) * sc;
    }
}

// ---------------------------------------------------------------------------
// K3: gather W1 column x[n], apply BN1 + exact GELU, store bf16 activations.
//     W1 (142.6 MB) fits in the 192 MB L2 and was just streamed by k_stats1.
// ---------------------------------------------------------------------------
__global__ __launch_bounds__(256) void k_gather_act(
    const float* __restrict__ W1, const int* __restrict__ x,
    const float* __restrict__ b1,
    const float* __restrict__ scale1, const float* __restrict__ shift1,
    __bf16* __restrict__ h1act)
{
    const int n   = blockIdx.x;          // 0..NB-1
    const int idx = x[n];
    __bf16* orow  = h1act + (size_t)n * INTER;
    for (int c = threadIdx.x; c < INTER; c += 256) {
        float h = W1[(size_t)c * VOCAB + idx] + b1[c];
        float y = fmaf(h, scale1[c], shift1[c]);
        orow[c] = (__bf16)gelu_exact(y);
    }
}

// K4: convert W2 to bf16
__global__ void k_cvt_w2(const float* __restrict__ W2, __bf16* __restrict__ W2b) {
    const int total = EMB * INTER;
    int stride = gridDim.x * blockDim.x;
    for (int i = blockIdx.x * blockDim.x + threadIdx.x; i < total; i += stride)
        W2b[i] = (__bf16)W2[i];
}

// ---------------------------------------------------------------------------
// K5: WMMA GEMM  h2[n,e] = sum_c h1act[n,c]*W2b[e,c] + b2[e]
//     One 16x32 output tile per wave32 (two independent accumulators ->
//     2x WMMA ILP, A fragment reused across both N-tiles).  K-loop over
//     INTER in steps of 32 using v_wmma_f32_16x16x32_bf16.
//
//     A-frag (16x32 bf16): lane l -> row M=l&15,
//       lanes 0-15: K = 0..15; lanes 16-31: K = 16..31 (contiguous, as B).
//     B-frag (32x16 bf16): lane l -> col N=l&15,
//       lanes 0-15: K = 0..15; lanes 16-31: K = 16..31 (contiguous).
//     D/C (16x16 f32): VGPR r, lanes 0-15 -> M=r, N=l; lanes 16-31 -> M=r+8.
// ---------------------------------------------------------------------------
__global__ __launch_bounds__(256) void k_gemm_wmma(
    const __bf16* __restrict__ A,    // [NB, INTER] bf16 activations
    const __bf16* __restrict__ B,    // [EMB, INTER] bf16 weights
    const float* __restrict__ b2,
    float* __restrict__ out)         // [NB, EMB] fp32 (pre-BN2 h2)
{
    const int lane = threadIdx.x & 31;
    const int wid  = threadIdx.x >> 5;
    const int tile = blockIdx.x * 8 + wid;       // 8192 wave-tiles total
    const int rowT = tile >> 4;                  // 512 row tiles (M=16)
    const int colP = tile & 15;                  // 16 col pairs  (N=32)
    const int rowbase = rowT << 4;
    const int colbase = colP << 5;

    const int l15 = lane & 15;
    const int hi  = lane >> 4;                   // 0 or 1
    const int kbA = hi << 4;                     // 0 or 16
    const int kbB = hi << 4;                     // 0 or 16

    const __bf16* aRow  = A + (size_t)(rowbase + l15) * INTER;
    const __bf16* bRow0 = B + (size_t)(colbase + l15) * INTER;
    const __bf16* bRow1 = B + (size_t)(colbase + 16 + l15) * INTER;

    v8f acc0 = {};
    v8f acc1 = {};
    for (int kt = 0; kt < INTER; kt += 32) {
        if (kt + 32 < INTER) {
            __builtin_prefetch(aRow  + kt + 32 + kbA, 0, 1); // global_prefetch_b8
            __builtin_prefetch(bRow0 + kt + 32 + kbB, 0, 1);
            __builtin_prefetch(bRow1 + kt + 32 + kbB, 0, 1);
        }
        v8bf alo = *(const v8bf*)(aRow + kt + kbA);          // K = kbA .. kbA+7
        v8bf ahi = *(const v8bf*)(aRow + kt + kbA + 8);      // K = kbA+8 .. kbA+15
        v16bf a  = __builtin_shufflevector(alo, ahi,
                      0, 1, 2, 3, 4, 5, 6, 7, 8, 9, 10, 11, 12, 13, 14, 15);
        v16bf b0 = *(const v16bf*)(bRow0 + kt + kbB);        // K = kbB .. kbB+15
        v16bf b1f = *(const v16bf*)(bRow1 + kt + kbB);
        acc0 = __builtin_amdgcn_wmma_f32_16x16x32_bf16(
                  false, a, false, b0,  (short)0, acc0, false, false);
        acc1 = __builtin_amdgcn_wmma_f32_16x16x32_bf16(
                  false, a, false, b1f, (short)0, acc1, false, false);
    }

    const float bb0 = b2[colbase + l15];
    const float bb1 = b2[colbase + 16 + l15];
    float* o0 = out + (size_t)(rowbase + (hi << 3)) * EMB + colbase + l15;
    #pragma unroll
    for (int r = 0; r < 8; ++r) {
        o0[(size_t)r * EMB]      = acc0[r] + bb0;
        o0[(size_t)r * EMB + 16] = acc1[r] + bb1;
    }
}

// ---------------------------------------------------------------------------
// K6: BN2 partial stats (deterministic: per-block partials, no float atomics)
//     64 blocks x 128 rows; thread t owns columns t and t+256.
// ---------------------------------------------------------------------------
__global__ __launch_bounds__(256) void k_stats2_partial(
    const float* __restrict__ h2,
    float* __restrict__ psum, float* __restrict__ psq)   // [64, 512] each
{
    const int tid  = threadIdx.x;
    const int base = blockIdx.x * 128;
    const int c0 = tid, c1 = tid + 256;
    float a0 = 0.f, a1 = 0.f, q0 = 0.f, q1 = 0.f;
    for (int r = 0; r < 128; ++r) {
        const float* row = h2 + (size_t)(base + r) * EMB;
        float v0 = row[c0]; a0 += v0; q0 += v0 * v0;
        float v1 = row[c1]; a1 += v1; q1 += v1 * v1;
    }
    float* ps = psum + (size_t)blockIdx.x * EMB;
    float* pq = psq  + (size_t)blockIdx.x * EMB;
    ps[c0] = a0; ps[c1] = a1;
    pq[c0] = q0; pq[c1] = q1;
}

// K7: finalize BN2 scale/shift
__global__ void k_finalize2(
    const float* __restrict__ psum, const float* __restrict__ psq,
    const float* __restrict__ g2, const float* __restrict__ beta2,
    float* __restrict__ scale2, float* __restrict__ shift2)
{
    int c = blockIdx.x * blockDim.x + threadIdx.x;
    if (c >= EMB) return;
    float s1 = 0.f, s2 = 0.f;
    for (int b = 0; b < 64; ++b) {
        s1 += psum[(size_t)b * EMB + c];
        s2 += psq [(size_t)b * EMB + c];
    }
    float invN = 1.0f / (float)NB;
    float m    = s1 * invN;
    float var  = s2 * invN - m * m;
    float sc   = g2[c] * rsqrtf(var + BN_EPS);
    scale2[c]  = sc;
    shift2[c]  = beta2[c] - m * sc;
}

// K8: BN2 + exact GELU, in-place on d_out (elementwise read->write)
__global__ void k_apply2(float* __restrict__ out,
                         const float* __restrict__ scale2,
                         const float* __restrict__ shift2)
{
    const int total = NB * EMB;
    int stride = gridDim.x * blockDim.x;
    for (int i = blockIdx.x * blockDim.x + threadIdx.x; i < total; i += stride) {
        int c   = i & (EMB - 1);
        float y = fmaf(out[i], scale2[c], shift2[c]);
        out[i]  = gelu_exact(y);
    }
}

// ---------------------------------------------------------------------------
// Launch
// ---------------------------------------------------------------------------
extern "C" void kernel_launch(void* const* d_in, const int* in_sizes, int n_in,
                              void* d_out, int out_size, void* d_ws, size_t ws_size,
                              hipStream_t stream) {
    (void)in_sizes; (void)n_in; (void)out_size; (void)ws_size;

    const int*   x_t   = (const int*)  d_in[0];
    const float* W1    = (const float*)d_in[1];
    const float* b1    = (const float*)d_in[2];
    const float* g1    = (const float*)d_in[3];
    const float* beta1 = (const float*)d_in[4];
    const float* W2    = (const float*)d_in[5];
    const float* b2    = (const float*)d_in[6];
    const float* g2    = (const float*)d_in[7];
    const float* beta2 = (const float*)d_in[8];
    float* out = (float*)d_out;          // doubles as h2 storage pre-BN2

    // workspace layout (all offsets 256B-aligned)
    char* ws = (char*)d_ws;
    unsigned int* cnt    = (unsigned int*)(ws + 0);            // 32 KB
    float*  scale1 = (float*) (ws + 32768);                    // 17408
    float*  shift1 = (float*) (ws + 50176);                    // 17408
    float*  scale2 = (float*) (ws + 67584);                    // 2048
    float*  shift2 = (float*) (ws + 69632);                    // 2048
    float*  psum   = (float*) (ws + 71680);                    // 64*512*4 = 128 KB
    float*  psq    = (float*) (ws + 202752);                   // 128 KB
    __bf16* W2b    = (__bf16*)(ws + 333824);                   // 4.46 MB
    __bf16* h1act  = (__bf16*)(ws + 4790272);                  // 71.3 MB

    k_zero          <<<VOCAB / 256, 256, 0, stream>>>(cnt);
    k_hist          <<<NB / 256,    256, 0, stream>>>(x_t, cnt);
    k_stats1        <<<INTER,       256, 0, stream>>>(W1, cnt, b1, g1, beta1,
                                                      scale1, shift1);
    k_gather_act    <<<NB,          256, 0, stream>>>(W1, x_t, b1, scale1, shift1,
                                                      h1act);
    k_cvt_w2        <<<2048,        256, 0, stream>>>(W2, W2b);
    k_gemm_wmma     <<<1024,        256, 0, stream>>>(h1act, W2b, b2, out);
    k_stats2_partial<<<64,          256, 0, stream>>>(out, psum, psq);
    k_finalize2     <<<2,           256, 0, stream>>>(psum, psq, g2, beta2,
                                                      scale2, shift2);
    k_apply2        <<<2048,        256, 0, stream>>>(out, scale2, shift2);
}
